// MLP_PG_22892175688409
// MI455X (gfx1250) — matrix operand order repaired and checked by reading, not compiled, not run on hardware
//
#include <hip/hip_runtime.h>
#include <math.h>

typedef __attribute__((ext_vector_type(16))) __bf16 v16bf;
typedef __attribute__((ext_vector_type(8)))  float  v8f;
typedef __attribute__((ext_vector_type(4)))  unsigned v4u;
typedef __attribute__((ext_vector_type(8)))  int      v8i;
typedef __attribute__((ext_vector_type(4)))  int      v4i;

#if __has_builtin(__builtin_amdgcn_tensor_load_to_lds) && __has_builtin(__builtin_amdgcn_s_wait_tensorcnt)
#define K3_TDM 1
#else
#define K3_TDM 0
#endif

// ---- workspace layout (dword offsets) ----
#define WS_SUMS   16        // 4 cats * 64 feats
#define WS_MEANS  288       // 7 * 64
#define WS_CNT4   768       // NB*4   (NB = N/256 <= 512)
#define WS_BASE4  2816      // NB*4
#define WS_BASE2  4864      // NB*2
#define WS_COUNTS 5888      // 4
#define WS_OTAB   5896      // 7
#define WS_W1BT   6144      // 1024*64 u32 (bf16-packed, tile-swizzled W1)
#define WS_RANK1  71680     // N
// WS_RANK2 = WS_RANK1 + N, WS_PART = WS_RANK1 + 2N, WS_PGP = WS_RANK1 + 4N

__device__ __forceinline__ unsigned short f2bf(float f){
  union { float f; unsigned u; } a; a.f = f;
  unsigned r = a.u + 0x7FFFu + ((a.u >> 16) & 1u);   // round-to-nearest-even
  return (unsigned short)(r >> 16);
}

// branchless SELU via native exp (v_exp_f32): scale*max(z,0) + scale*alpha*(exp(min(z,0))-1)
__device__ __forceinline__ float selu_f(float z){
  float zp = fmaxf(z, 0.f), zn = fminf(z, 0.f);
  return 1.0507009873554805f * zp + 1.7580993408473766f * (__expf(zn) - 1.f);
}

// ---- k0: repack W1 (128x1024 f32) into bf16 per-lane WMMA-B tile images.
// Tile j dword d = t*256 + v*32 + l holds B elements (K = 32t + 16h + 2v + {0,1},
// col = j*16 + (l&15)), h = l>>4 — contiguous k order, same as the A fragments.
__global__ void k0_prep_w1(const float* __restrict__ W1, unsigned* __restrict__ w1bt){
  int idx = blockIdx.x * 256 + threadIdx.x;           // 0..65535
  int j = idx >> 10, r = idx & 1023;
  int t = r >> 8, v = (r >> 5) & 7, l = r & 31;
  int h = l >> 4, m = l & 15;
  int K = 32*t + 16*h + 2*v;
  int col = j*16 + m;
  unsigned lo = f2bf(W1[(size_t)K * 1024 + col]);
  unsigned hi = f2bf(W1[(size_t)(K + 1) * 1024 + col]);
  w1bt[idx] = lo | (hi << 16);
}

// ---- k1: per-block partial segment sums over 4 leaf categories (deterministic)
__global__ void k1_segsum(const float* __restrict__ x, const int* __restrict__ s0p,
                          const int* __restrict__ s1p, float* __restrict__ part){
  int tid = threadIdx.x, f = tid & 63, g = tid >> 6;
  int base = blockIdx.x * 128;
  float a0 = 0.f, a1 = 0.f, a2 = 0.f, a3 = 0.f;
  for (int i = 0; i < 32; ++i){
    int row = base + g + 4*i;
    int cat = 2*s0p[row] + s1p[row];
    float xv = x[(size_t)row * 64 + f];
    a0 += (cat == 0) ? xv : 0.f;
    a1 += (cat == 1) ? xv : 0.f;
    a2 += (cat == 2) ? xv : 0.f;
    a3 += (cat == 3) ? xv : 0.f;
  }
  __shared__ float sd[1024];
  sd[g*256 +   0 + f] = a0; sd[g*256 +  64 + f] = a1;
  sd[g*256 + 128 + f] = a2; sd[g*256 + 192 + f] = a3;
  __syncthreads();
  if (tid < 64){
    for (int c = 0; c < 4; ++c)
      part[(size_t)blockIdx.x * 256 + c*64 + tid] =
        sd[c*64 + tid] + sd[256 + c*64 + tid] + sd[512 + c*64 + tid] + sd[768 + c*64 + tid];
  }
}

__global__ void k1b_reduce(const float* __restrict__ part, float* __restrict__ sums, int nb){
  int tid = threadIdx.x;                               // 256 = 4 cats * 64 feats
  float s = 0.f;
  for (int b = 0; b < nb; ++b) s += part[(size_t)b * 256 + tid];
  sums[tid] = s;
}

// ---- r1: per-block 4-category histograms (one row per thread)
__global__ void r1_count(const int* __restrict__ s0p, const int* __restrict__ s1p,
                         int* __restrict__ cnt4){
  __shared__ int h[4];
  int tid = threadIdx.x;
  if (tid < 4) h[tid] = 0;
  __syncthreads();
  int row = blockIdx.x * 256 + tid;
  atomicAdd(&h[2*s0p[row] + s1p[row]], 1);
  __syncthreads();
  if (tid < 4) cnt4[blockIdx.x * 4 + tid] = h[tid];
}

// ---- r2: exclusive scans of block histograms + totals + edge-block offset table
__global__ void r2_scan(const int* __restrict__ cnt4, int* __restrict__ base4,
                        int* __restrict__ base2, int* __restrict__ counts,
                        int* __restrict__ otab, int NB, int N){
  int tid = threadIdx.x;
  if (tid < 4){
    int run = 0;
    for (int b = 0; b < NB; ++b){ base4[b*4 + tid] = run; run += cnt4[b*4 + tid]; }
    counts[tid] = run;
  } else if (tid < 6){
    int c = tid - 4, run = 0;
    for (int b = 0; b < NB; ++b){ base2[b*2 + c] = run; run += cnt4[b*4 + 2*c] + cnt4[b*4 + 2*c + 1]; }
  }
  __syncthreads();
  if (tid == 0){
    int c00 = counts[0], c01 = counts[1], c10 = counts[2], c11 = counts[3];
    int c0 = c00 + c01, c1 = c10 + c11;
    otab[0] = 0;
    otab[1] = 2*(N + 1);
    otab[2] = otab[1] + 2*(c0 + 2);
    otab[3] = otab[2] + 2*(c1 + 2);
    otab[4] = otab[3] + 2*(c00 + 3);
    otab[5] = otab[4] + 2*(c01 + 3);
    otab[6] = otab[5] + 2*(c10 + 3);
  }
}

// ---- r3: per-row ranks within depth-1 (s0) and depth-2 (s0,s1) segments
__global__ void r3_ranks(const int* __restrict__ s0p, const int* __restrict__ s1p,
                         const int* __restrict__ base4, const int* __restrict__ base2,
                         int* __restrict__ rank1, int* __restrict__ rank2){
  __shared__ unsigned char cats[256];
  int tid = threadIdx.x;
  int row = blockIdx.x * 256 + tid;
  int cat = 2*s0p[row] + s1p[row];
  cats[tid] = (unsigned char)cat;
  __syncthreads();
  int lr1 = 0, lr2 = 0, p = cat >> 1;
  for (int j = 0; j < tid; ++j){
    int c = cats[j];
    lr2 += (c == cat);
    lr1 += ((c >> 1) == p);
  }
  rank2[row] = base4[blockIdx.x*4 + cat] + lr2;
  rank1[row] = base2[blockIdx.x*2 + p] + lr1;
}

// ---- k2: means, augmented x_upd rows, node<->node + self edge values
__global__ void k2_finalize(const float* __restrict__ sums, const int* __restrict__ counts,
                            const int* __restrict__ otab, float* __restrict__ means_ws,
                            float* __restrict__ d_out, int N){
  __shared__ float sm[7*64];
  int tid = threadIdx.x;
  if (tid < 64){
    int f = tid;
    float s0 = sums[f], s1 = sums[64+f], s2 = sums[128+f], s3 = sums[192+f];
    float c00 = (float)counts[0], c01 = (float)counts[1];
    float c10 = (float)counts[2], c11 = (float)counts[3];
    float m[7];
    m[0] = (s0 + s1 + s2 + s3) / (float)N;
    m[1] = (s0 + s1) / (c00 + c01);
    m[2] = (s2 + s3) / (c10 + c11);
    m[3] = s0 / c00; m[4] = s1 / c01; m[5] = s2 / c10; m[6] = s3 / c11;
    for (int nd = 0; nd < 7; ++nd){
      sm[nd*64 + f] = m[nd];
      means_ws[nd*64 + f] = m[nd];
      d_out[(size_t)(N + nd) * 65 + f] = m[nd];
    }
  } else if (tid < 71){
    d_out[(size_t)(N + tid - 64) * 65 + 64] = 1.0f;   // indicator of augmented nodes
  }
  __syncthreads();
  if (tid == 0){
    float* v = d_out + (size_t)(N + 7) * 65;
    int c00 = counts[0], c01 = counts[1], c10 = counts[2], c11 = counts[3];
    int c0 = c00 + c01, c1 = c10 + c11;
    v[N] = 1.0f; v[2*N + 1] = 1.0f;                    // depth-0 self edge (both dirs)
    for (int s = 0; s < 2; ++s){
      int base = otab[1 + s], cs = s ? c1 : c0, L = cs + 2;
      float d2 = 0.f;
      for (int f = 0; f < 64; ++f){ float d = sm[f] - sm[(1+s)*64 + f]; d2 += d*d; }
      float e0 = __expf(-d2);
      v[base + cs] = e0;        v[base + cs + 1] = 1.0f;
      v[base + L + cs] = e0;    v[base + L + cs + 1] = 1.0f;
    }
    int cc4[4] = {c00, c01, c10, c11};
    for (int cat = 0; cat < 4; ++cat){
      int base = otab[3 + cat], c = cc4[cat], L = c + 3, s0 = cat >> 1;
      float d20 = 0.f, d21 = 0.f;
      for (int f = 0; f < 64; ++f){
        float a = sm[(3+cat)*64 + f];
        float d0 = sm[f] - a;             d20 += d0*d0;
        float d1 = sm[(1+s0)*64 + f] - a; d21 += d1*d1;
      }
      float e0 = __expf(-d20), e1 = __expf(-d21);
      v[base + c] = e0;     v[base + c + 1] = e1;     v[base + c + 2] = 1.0f;
      v[base + L + c] = e0; v[base + L + c + 1] = e1; v[base + L + c + 2] = 1.0f;
    }
  }
}

// ---- k3: fused (concat -> GEMM(bf16 WMMA) -> bias -> SELU -> dot W2 -> sigmoid -> pg)
// 8 waves/block, 16 rows per wave. W1 tiles streamed via TDM (tensor_load_to_lds)
// into a double-buffered LDS with one barrier per tile; fallback = register pipeline.
__global__ void __launch_bounds__(256) k3_mlp(
    const float* __restrict__ x, const unsigned* __restrict__ w1bt,
    const float* __restrict__ b1, const float* __restrict__ W2, const float* __restrict__ b2,
    const int* __restrict__ s0p, const int* __restrict__ s1p, const int* __restrict__ s2p,
    const float* __restrict__ means, float* __restrict__ pgp, int N){
  __shared__ unsigned lds[2048];                       // two 128x16 bf16 B tiles
  const int tid  = threadIdx.x;
  const int lane = tid & 31;
  const int wave = tid >> 5;
  const int m    = lane & 15;
  const int h    = lane >> 4;
  const int depth = blockIdx.y;
  const int rowBase = (blockIdx.x * 8 + wave) * 16;
  const int row = rowBase + m;
  const int s0 = s0p[row], s1 = s1p[row];
  const int node = (depth == 0) ? 0 : ((depth == 1) ? (1 + s0) : (3 + 2*s0 + s1));
  const float* xr = x + (size_t)row * 64;
  const float* mr = means + node * 64;

  // A fragments: lane holds K run [32t+16h,+16) (contiguous, matches B)
  union AF { v16bf v; unsigned short e[16]; } a[4];
  #pragma unroll
  for (int t = 0; t < 4; ++t){
    int k0 = 32*t + 16*h;
    #pragma unroll
    for (int e = 0; e < 8; ++e){
      int ka = k0 + e, kb = k0 + 8 + e;
      float fa = (ka < 64) ? xr[ka] : mr[ka - 64];
      float fb = (kb < 64) ? xr[kb] : mr[kb - 64];
      a[t].e[e]     = f2bf(fa);
      a[t].e[8 + e] = f2bf(fb);
    }
  }

#if K3_TDM
  const unsigned ldsbase = (unsigned)(unsigned long long)(void*)&lds[0];
  const unsigned long long gbase = (unsigned long long)(const void*)w1bt;
  auto tdm_issue = [&](int j, int buf){
    unsigned long long ga = gbase + (unsigned long long)j * 4096ull;
    v4u g0;
    g0.x = 1u;                                        // count = 1 valid descriptor
    g0.y = ldsbase + (unsigned)buf * 4096u;           // LDS byte address
    g0.z = (unsigned)(ga & 0xFFFFFFFFull);            // global addr [31:0]
    g0.w = (unsigned)((ga >> 32) & 0x01FFFFFFull) | 0x80000000u;  // addr[56:32] | type=2
    v8i g1;
    g1[0] = 0x20000;      // data_size = 4B
    g1[1] = 0x04000000;   // tensor_dim0 = 1024 (low16 at bits[63:48])
    g1[2] = 0x00010000;   // tensor_dim1 = 1
    g1[3] = 0x04000000;   // tile_dim0 = 1024 (bits[127:112])
    g1[4] = 0x00000001;   // tile_dim1 = 1
    g1[5] = 0x00000400;   // tensor_dim0_stride = 1024
    g1[6] = 0; g1[7] = 0;
    v4i z4 = {0, 0, 0, 0};
#if __clang_major__ >= 23
    v8i z8 = {0, 0, 0, 0, 0, 0, 0, 0};
    __builtin_amdgcn_tensor_load_to_lds(g0, g1, z4, z4, z8, 0);   // clang-23 6-arg form
#else
    __builtin_amdgcn_tensor_load_to_lds(g0, g1, z4, z4, 0);       // ROCm 7.2 5-arg form
#endif
  };
  if (wave == 0) tdm_issue(0, 0);                     // prefetch tile 0 -> buf0
#else
  const uint4* w1v = (const uint4*)w1bt;
  ((uint4*)lds)[tid] = w1v[tid];                      // tile 0 -> buf0
#endif

  float acc[8] = {0.f,0.f,0.f,0.f,0.f,0.f,0.f,0.f};
  for (int j = 0; j < 64; ++j){
    int cur = j & 1;
#if K3_TDM
    __builtin_amdgcn_s_wait_tensorcnt(0);             // tile j landed (no-op for waves 1..7)
    __syncthreads();                                  // publish buf[cur] to all waves
    if (j + 1 < 64 && wave == 0) tdm_issue(j + 1, cur ^ 1);  // DMA next tile during compute
#else
    __syncthreads();
    uint4 nxt;
    if (j + 1 < 64) nxt = w1v[(j + 1) * 256 + tid];   // long-latency load overlaps compute
#endif
    const unsigned* bufp = lds + cur * 1024;
    // hoist the whole 128x16 B tile (4 fragments) -> single DS wait per tile
    union BF { v16bf v; unsigned u[8]; } b[4];
    #pragma unroll
    for (int t = 0; t < 4; ++t)
      #pragma unroll
      for (int vv = 0; vv < 8; ++vv) b[t].u[vv] = bufp[t*256 + vv*32 + lane];  // conflict-free
    // two independent accumulators break the WMMA C-chain RAW hazard
    v8f c0 = {0.f,0.f,0.f,0.f,0.f,0.f,0.f,0.f};
    v8f c1 = {0.f,0.f,0.f,0.f,0.f,0.f,0.f,0.f};
    c0 = __builtin_amdgcn_wmma_f32_16x16x32_bf16(false, a[0].v, false, b[0].v, (short)0, c0, false, false);
    c1 = __builtin_amdgcn_wmma_f32_16x16x32_bf16(false, a[1].v, false, b[1].v, (short)0, c1, false, false);
    c0 = __builtin_amdgcn_wmma_f32_16x16x32_bf16(false, a[2].v, false, b[2].v, (short)0, c0, false, false);
    c1 = __builtin_amdgcn_wmma_f32_16x16x32_bf16(false, a[3].v, false, b[3].v, (short)0, c1, false, false);
    float bb = b1[j*16 + m];
    float wv = W2[j*16 + m];
    #pragma unroll
    for (int r = 0; r < 8; ++r)
      acc[r] += selu_f(c0[r] + c1[r] + bb) * wv;      // branchless, native exp
#if !K3_TDM
    if (j + 1 < 64) ((uint4*)(lds + (cur ^ 1) * 1024))[tid] = nxt;
#endif
  }

  // reduce the 16 column-lanes of each half-wave (rows r / r+8)
  #pragma unroll
  for (int r = 0; r < 8; ++r){
    float v = acc[r];
    v += __shfl_xor(v, 1); v += __shfl_xor(v, 2);
    v += __shfl_xor(v, 4); v += __shfl_xor(v, 8);
    acc[r] = v;
  }
  if (m == 0){
    const int* sd = (depth == 0) ? s0p : ((depth == 1) ? s1p : s2p);
    float b2v = b2[0];
    float tot = 0.f;
    int rb = rowBase + 8*h;
    #pragma unroll
    for (int r = 0; r < 8; ++r){
      float p = 1.f / (1.f + __expf(-(acc[r] + b2v)));
      float s = (float)sd[rb + r];
      tot += s*p + (1.f - s)*(1.f - p);
    }
    pgp[((size_t)depth * (N >> 4) + blockIdx.x * 8 + wave) * 2 + h] = tot;
  }
}

// ---- k4: x_upd copy + per-row gaussian edge values at rank-derived positions
__global__ void __launch_bounds__(128) k4_edges(
    const float* __restrict__ x, const float* __restrict__ means,
    const int* __restrict__ s0p, const int* __restrict__ s1p,
    const int* __restrict__ rank1, const int* __restrict__ rank2,
    const int* __restrict__ counts, const int* __restrict__ otab,
    float* __restrict__ d_out, int N){
  __shared__ float sx[128 * 65];
  __shared__ float sm[7 * 64];
  int tid = threadIdx.x;
  int base = blockIdx.x * 128;
  for (int i = tid; i < 448; i += 128) sm[i] = means[i];
  for (int i = tid; i < 128 * 64; i += 128){
    int r = i >> 6, f = i & 63;
    float val = x[(size_t)(base + r) * 64 + f];
    sx[r*65 + f] = val;
    d_out[(size_t)(base + r) * 65 + f] = val;
  }
  d_out[(size_t)(base + tid) * 65 + 64] = 0.0f;        // indicator of original rows
  __syncthreads();
  int row = base + tid;
  int s0 = s0p[row], s1 = s1p[row], cat = 2*s0 + s1;
  const float* mA = &sm[0];
  const float* mB = &sm[(1 + s0) * 64];
  const float* mC = &sm[(3 + cat) * 64];
  float dA = 0.f, dB = 0.f, dC = 0.f;
  for (int f = 0; f < 64; ++f){
    float xv = sx[tid*65 + f];
    float da = xv - mA[f]; dA += da*da;
    float db = xv - mB[f]; dB += db*db;
    float dc = xv - mC[f]; dC += dc*dc;
  }
  float v0 = __expf(-(dA + 1.f));                      // +1 = indicator mismatch
  float v1 = __expf(-(dB + 1.f));
  float v2 = __expf(-(dC + 1.f));
  float* v = d_out + (size_t)(N + 7) * 65;
  v[row] = v0; v[(N + 1) + row] = v0;
  int c0 = counts[0] + counts[1], c1 = counts[2] + counts[3];
  int r1 = rank1[row], r2 = rank2[row];
  int bo1 = otab[1 + s0], cs = s0 ? c1 : c0;
  v[bo1 + r1] = v1; v[bo1 + cs + 2 + r1] = v1;
  int bo2 = otab[3 + cat], cc = counts[cat];
  v[bo2 + r2] = v2; v[bo2 + cc + 3 + r2] = v2;
}

// ---- k5: fixed-order pg_sum reduction
__global__ void k5_pg(const float* __restrict__ pgp, float* __restrict__ d_out,
                      int M, size_t off){
  __shared__ float sd[256];
  int tid = threadIdx.x;
  float s = 0.f;
  for (int i = tid; i < M; i += 256) s += pgp[i];
  sd[tid] = s;
  __syncthreads();
  if (tid == 0){
    float t = 0.f;
    for (int i = 0; i < 256; ++i) t += sd[i];
    d_out[off] = t;
  }
}

extern "C" void kernel_launch(void* const* d_in, const int* in_sizes, int n_in,
                              void* d_out, int out_size, void* d_ws, size_t ws_size,
                              hipStream_t stream){
  const float* x  = (const float*)d_in[0];
  const float* W1 = (const float*)d_in[1];
  const float* b1 = (const float*)d_in[2];
  const float* W2 = (const float*)d_in[3];
  const float* b2 = (const float*)d_in[4];
  const int*   s0 = (const int*)d_in[5];
  const int*   s1 = (const int*)d_in[6];
  const int*   s2 = (const int*)d_in[7];
  float* out = (float*)d_out;
  const int N  = in_sizes[0] / 64;                     // F = 64
  const int NB = N / 256;

  float*    ws     = (float*)d_ws;
  float*    sums   = ws + WS_SUMS;
  float*    means  = ws + WS_MEANS;
  int*      cnt4   = (int*)d_ws + WS_CNT4;
  int*      base4  = (int*)d_ws + WS_BASE4;
  int*      base2  = (int*)d_ws + WS_BASE2;
  int*      counts = (int*)d_ws + WS_COUNTS;
  int*      otab   = (int*)d_ws + WS_OTAB;
  unsigned* w1bt   = (unsigned*)d_ws + WS_W1BT;
  int*      rank1  = (int*)d_ws + WS_RANK1;
  int*      rank2  = (int*)d_ws + WS_RANK1 + N;
  float*    part   = ws + WS_RANK1 + 2*N;              // (N/128) * 256
  float*    pgp    = ws + WS_RANK1 + 4*N;              // 3 * (N/16) * 2

  k0_prep_w1<<<256, 256, 0, stream>>>(W1, w1bt);
  k1_segsum<<<N/128, 256, 0, stream>>>(x, s0, s1, part);
  k1b_reduce<<<1, 256, 0, stream>>>(part, sums, N/128);
  r1_count<<<NB, 256, 0, stream>>>(s0, s1, cnt4);
  r2_scan<<<1, 64, 0, stream>>>(cnt4, base4, base2, counts, otab, NB, N);
  k2_finalize<<<1, 256, 0, stream>>>(sums, counts, otab, means, out, N);
  r3_ranks<<<NB, 256, 0, stream>>>(s0, s1, base4, base2, rank1, rank2);
  dim3 g3(N/128, 3);
  k3_mlp<<<g3, 256, 0, stream>>>(x, w1bt, b1, W2, b2, s0, s1, s2, means, pgp, N);
  k4_edges<<<N/128, 128, 0, stream>>>(x, means, s0, s1, rank1, rank2, counts, otab, out, N);
  k5_pg<<<1, 256, 0, stream>>>(pgp, out, 3*(N/16)*2, (size_t)out_size - 1);
}
